// PatchedSelfAttention_49185965474550
// MI455X (gfx1250) — matrix operand order repaired and checked by reading, not compiled, not run on hardware
//
#include <hip/hip_runtime.h>
#include <hip/hip_bf16.h>

// ---------------------------------------------------------------------------
// ChatGLM patched self-attention for MI455X (gfx1250, wave32, WMMA + TDM).
//  * all matmuls: v_wmma_f32_16x16x32_bf16, f32 accumulation
//  * operands converted to bf16 ONCE (bf16 weight set fits the 192MB L2),
//    weights stored pre-transposed [N][K] -> GEMM A/B tiles share one TDM shape
//  * LDS tiles DMA'd by the Tensor Data Mover, DOUBLE-BUFFERED: tile t+1 is
//    issued before waiting on tile t (s_wait_tensorcnt 2), overlapping DMA
//    with WMMA compute; descriptor-level row padding gives bank-safe layouts
//  * all WMMA fragments load as contiguous ds_load_b128 pairs
// ---------------------------------------------------------------------------

typedef __bf16 bf16_t;
typedef __attribute__((ext_vector_type(16))) __bf16    v16bf;
typedef __attribute__((ext_vector_type(8)))  __bf16    v8bf;
typedef __attribute__((ext_vector_type(8)))  float     v8f;
typedef __attribute__((ext_vector_type(4)))  unsigned  u32x4;
typedef __attribute__((ext_vector_type(8)))  int       i32x8;
typedef __attribute__((ext_vector_type(4)))  int       i32x4;

#define NS    4
#define SL    1024
#define HID   4096
#define NH    32
#define HD    128
#define NKV   2
#define BS    64
#define STOK  (NS * SL)                       // 4096 tokens
#define QKV_N (NH * HD + 2 * NKV * HD)        // 4608
#define GQ    (NH / NKV)                      // 16
#define ROT   (HD / 2)                        // 64

#if __has_builtin(__builtin_amdgcn_tensor_load_to_lds)
#define HAVE_TDM 1
#else
#define HAVE_TDM 0
#endif

// f32 -> bf16 (round-to-nearest via +0x8000), packed pair with one v_perm_b32
static __device__ __forceinline__ unsigned pack2(float a, float b) {
  unsigned ua = __builtin_bit_cast(unsigned, a) + 0x8000u;
  unsigned ub = __builtin_bit_cast(unsigned, b) + 0x8000u;
  return __builtin_amdgcn_perm(ub, ua, 0x07060302u);  // {b.hi16, a.hi16}
}
static __device__ __forceinline__ bf16_t f2bf(float f) {
  unsigned short h = (unsigned short)((__builtin_bit_cast(unsigned, f) + 0x8000u) >> 16);
  return __builtin_bit_cast(bf16_t, h);
}
// two contiguous 16B loads -> one 16-element bf16 fragment
static __device__ __forceinline__ v16bf ld2x8(const bf16_t* p0, const bf16_t* p1) {
  v8bf a = *(const v8bf*)p0;
  v8bf b = *(const v8bf*)p1;
  return __builtin_shufflevector(a, b, 0, 1, 2, 3, 4, 5, 6, 7,
                                 8, 9, 10, 11, 12, 13, 14, 15);
}

#if HAVE_TDM
// Issue one TDM 2D tile load (ISA ch.8 D# layout). Call from ONE wave only
// (wave-uniform guard; EXEC ignored by tensor ops). Tracked by TENSORcnt,
// completes in-order with other tensor ops of the same wave.
static __device__ __forceinline__ void tdm_load_2d(
    const void* gaddr, unsigned lds_off, unsigned tile_w, unsigned tile_h,
    unsigned stride0, unsigned tensor_w, unsigned tensor_h,
    unsigned pad_code, unsigned pad_amt_dw) {
  unsigned long long ga = (unsigned long long)(uintptr_t)gaddr;
  u32x4 g0;
  g0[0] = 1u;                                        // count=1, user mode
  g0[1] = lds_off;                                   // LDS byte offset
  g0[2] = (unsigned)(ga & 0xffffffffu);              // global_addr[31:0]
  g0[3] = (unsigned)((ga >> 32) & 0x01ffffffu) | (2u << 30);  // [56:32] | type=2
  unsigned pe = pad_amt_dw ? 1u : 0u;
  unsigned pa = pad_amt_dw ? (pad_amt_dw - 1u) : 0u;
  unsigned long long s0 = (unsigned long long)stride0;
  i32x8 g1;
  g1[0] = (int)((1u << 16) | (pe << 20) | (pad_code << 22) | (pa << 25));
  g1[1] = (int)((tensor_w & 0xffffu) << 16);                     // dim0[15:0]
  g1[2] = (int)((tensor_w >> 16) | ((tensor_h & 0xffffu) << 16));
  g1[3] = (int)((tensor_h >> 16) | (tile_w << 16));              // tile_dim0
  g1[4] = (int)(tile_h & 0xffffu);                               // tile_dim1
  g1[5] = (int)(s0 & 0xffffffffu);
  g1[6] = (int)((s0 >> 32) & 0xffffu);                           // stride1=0
  g1[7] = 0;
  i32x4 z4 = {0, 0, 0, 0};
#if defined(__clang_major__) && __clang_major__ >= 23
  i32x8 z8 = {0, 0, 0, 0, 0, 0, 0, 0};
  __builtin_amdgcn_tensor_load_to_lds(g0, g1, z4, z4, z8, 0);
#else
  __builtin_amdgcn_tensor_load_to_lds(g0, g1, z4, z4, 0);
#endif
}
#endif  // HAVE_TDM

// ---------------------------------------------------------------------------
// One-shot converts: f32 -> bf16 (plain) and f32 [K][N] -> bf16 [N][K].
// ---------------------------------------------------------------------------
__global__ __launch_bounds__(256)
void conv_bf16(const float* __restrict__ src, bf16_t* __restrict__ dst) {
  size_t i = ((size_t)blockIdx.x * 256 + threadIdx.x) * 4;
  float4 v = *(const float4*)(src + i);
  *(uint2*)(dst + i) = make_uint2(pack2(v.x, v.y), pack2(v.z, v.w));
}

__global__ __launch_bounds__(256)
void conv_bf16_T(const float* __restrict__ src, bf16_t* __restrict__ dst,
                 int K, int N) {  // src [K][N] -> dst [N][K]
  __shared__ float t[32][65];
  const int kb = blockIdx.y * 32, nb = blockIdx.x * 64;
  {
    int r = threadIdx.x >> 3, c = (threadIdx.x & 7) * 8;
    const float* s = src + (size_t)(kb + r) * N + nb + c;
#pragma unroll
    for (int i = 0; i < 8; ++i) t[r][c + i] = s[i];
  }
  __syncthreads();
  int n = threadIdx.x >> 2, kc = (threadIdx.x & 3) * 8;
  unsigned p0 = pack2(t[kc + 0][n], t[kc + 1][n]);
  unsigned p1 = pack2(t[kc + 2][n], t[kc + 3][n]);
  unsigned p2 = pack2(t[kc + 4][n], t[kc + 5][n]);
  unsigned p3 = pack2(t[kc + 6][n], t[kc + 7][n]);
  *(uint4*)(dst + (size_t)(nb + n) * K + kb + kc) = make_uint4(p0, p1, p2, p3);
}

// ---------------------------------------------------------------------------
// GEMM: C[M,N] = A[M,K] * BT[N,K]^T (+bias). WG = 8 waves, 128x128 tile.
// Double-buffered [128 rows][32 k] LDS tiles (80B padded rows) streamed by
// TDM: issue tile t+1, wait tensorcnt<=2 (tile t ready), compute, flip.
// Fragments (ISA 7.12.2): A lane=M, k runs [8h..],[16+8h..]; B lane=N,
// k runs [8h..],[16+8h..]; C lane=N, row r+8h.
// ---------------------------------------------------------------------------
__global__ __launch_bounds__(256)
void gemm_bf16wmma(const bf16_t* __restrict__ A, const bf16_t* __restrict__ BT,
                   const float* __restrict__ bias, float* __restrict__ C,
                   int M, int N, int K) {
  __shared__ bf16_t As[2][128][40];
  __shared__ bf16_t BsT[2][128][40];
  const int tid  = threadIdx.x;
  const int lane = tid & 31;
  const int wv   = tid >> 5;
  const int half = lane >> 4;
  const int l16  = lane & 15;
  const int n0   = blockIdx.x * 128;
  const int m0   = blockIdx.y * 128;
  const int nt   = K / 32;

  v8f acc[8];
  for (int c = 0; c < 8; ++c)
    for (int r = 0; r < 8; ++r) acc[c][r] = 0.0f;

#if HAVE_TDM
  if (wv == 0) {  // prologue: tile 0 in flight
    tdm_load_2d(A + (size_t)m0 * K, (unsigned)(uintptr_t)&As[0][0][0],
                32, 128, K, K, M, /*pad 16dw*/3, /*+4dw*/4);
    tdm_load_2d(BT + (size_t)n0 * K, (unsigned)(uintptr_t)&BsT[0][0][0],
                32, 128, K, K, N, 3, 4);
  }
#endif
  int cur = 0;
  for (int t = 0; t < nt; ++t) {
    const int k0 = t * 32;
#if HAVE_TDM
    if (wv == 0) {
      if (t + 1 < nt) {  // overlap: DMA tile t+1 while computing tile t
        tdm_load_2d(A + (size_t)m0 * K + k0 + 32,
                    (unsigned)(uintptr_t)&As[1 - cur][0][0], 32, 128, K, K, M, 3, 4);
        tdm_load_2d(BT + (size_t)n0 * K + k0 + 32,
                    (unsigned)(uintptr_t)&BsT[1 - cur][0][0], 32, 128, K, K, N, 3, 4);
        __builtin_amdgcn_s_wait_tensorcnt(2);  // tile t done, t+1 in flight
      } else {
        __builtin_amdgcn_s_wait_tensorcnt(0);
      }
    }
    __syncthreads();
#else
    for (int i = tid; i < 512; i += 256) {
      int r = i >> 2, c = (i & 3) * 8;
      *(uint4*)&As[cur][r][c]  = *(const uint4*)(A  + (size_t)(m0 + r) * K + k0 + c);
      *(uint4*)&BsT[cur][r][c] = *(const uint4*)(BT + (size_t)(n0 + r) * K + k0 + c);
    }
    __syncthreads();
#endif

    const bf16_t* arow = &As[cur][wv * 16 + l16][0];
    v16bf af = ld2x8(arow + 8 * half, arow + 16 + 8 * half);
#pragma unroll
    for (int c = 0; c < 8; ++c) {
      const bf16_t* brow = &BsT[cur][16 * c + l16][0];
      v16bf bfr = ld2x8(brow + 8 * half, brow + 16 + 8 * half);
      acc[c] = __builtin_amdgcn_wmma_f32_16x16x32_bf16(
          false, af, false, bfr, (short)0, acc[c], false, false);
    }
    __syncthreads();  // reads done before this buffer is DMA-overwritten
    cur ^= 1;
  }

#pragma unroll
  for (int c = 0; c < 8; ++c) {
    int col  = n0 + 16 * c + l16;
    float bv = bias ? bias[col] : 0.0f;
#pragma unroll
    for (int r = 0; r < 8; ++r) {
      int row = m0 + wv * 16 + r + 8 * half;
      C[(size_t)row * N + col] = acc[c][r] + bv;
    }
  }
}

// ---------------------------------------------------------------------------
// RoPE + paged KV scatter: Q [tok][NH*HD] bf16, K cache [slot][NKV*HD] bf16,
// V cache TRANSPOSED [g][dim][slot] bf16 (contraction-contiguous for PV).
// ---------------------------------------------------------------------------
__global__ __launch_bounds__(256)
void rope_scatter(const float* __restrict__ qkv, const float* __restrict__ rope,
                  const int* __restrict__ block_offsets,
                  bf16_t* __restrict__ qbuf, bf16_t* __restrict__ kbuf,
                  bf16_t* __restrict__ vt) {
  const int tok = blockIdx.x;
  const int tid = threadIdx.x;
  __shared__ float cs[ROT];
  if (tid < ROT) cs[tid] = rope[(size_t)tok * ROT + tid];
  __syncthreads();

  const float* q = qkv + (size_t)tok * QKV_N;
  const float* k = q + NH * HD;
  const float* v = k + NKV * HD;

  for (int idx = tid; idx < NH * (HD / 2); idx += 256) {
    int head = idx >> 6, pd = idx & 63, d0 = 2 * pd;
    float x0 = q[head * HD + d0], x1 = q[head * HD + d0 + 1];
    float o0 = x0, o1 = x1;
    if (pd < 32) {
      float c = cs[d0], s = cs[d0 + 1];
      o0 = x0 * c - x1 * s;
      o1 = x1 * c + x0 * s;
    }
    *(unsigned*)(qbuf + (size_t)tok * (NH * HD) + head * HD + d0) = pack2(o0, o1);
  }

  const int pos  = tok & (SL - 1);
  const int sid  = tok >> 10;
  const int slot = block_offsets[sid * (SL / BS) + (pos / BS)] * BS + (pos & (BS - 1));

  for (int idx = tid; idx < NKV * (HD / 2); idx += 256) {
    int g = idx >> 6, pd = idx & 63, d0 = 2 * pd;
    float x0 = k[g * HD + d0], x1 = k[g * HD + d0 + 1];
    float o0 = x0, o1 = x1;
    if (pd < 32) {
      float c = cs[d0], s = cs[d0 + 1];
      o0 = x0 * c - x1 * s;
      o1 = x1 * c + x0 * s;
    }
    *(unsigned*)(kbuf + (size_t)slot * (NKV * HD) + g * HD + d0) = pack2(o0, o1);
  }
  {
    int g = tid >> 7, d = tid & 127;  // 256 threads == NKV*HD
    vt[((size_t)g * HD + d) * STOK + slot] = f2bf(v[g * HD + d]);
  }
}

// ---------------------------------------------------------------------------
// Flash-style causal GQA attention; K/V tiles double-buffered through TDM;
// online softmax via width-16 __shfl_xor (a C-fragment row spans 16 lanes);
// P re-laid-out through padded LDS; ctx written directly as bf16.
// ---------------------------------------------------------------------------
__global__ __launch_bounds__(256)
void attention(const bf16_t* __restrict__ qbuf, const bf16_t* __restrict__ kbuf,
               const bf16_t* __restrict__ vt, const int* __restrict__ block_offsets,
               bf16_t* __restrict__ ctxb) {
  __shared__ bf16_t Kt[2][32][136];    // [key][feat], 272B rows
  __shared__ bf16_t VtT[2][128][40];   // [dim][key], 80B rows
  __shared__ bf16_t Pt[8][16][40];     // per-wave P, [q-row][key]
  const int tid  = threadIdx.x;
  const int lane = tid & 31;
  const int wv   = tid >> 5;
  const int half = lane >> 4;
  const int l16  = lane & 15;

  const int qb = blockIdx.x & 7;
  const int h  = (blockIdx.x >> 3) & 31;
  const int s  = blockIdx.x >> 8;
  const int g  = h / GQ;
  const int q0 = qb * 128;
  const int qrow0 = q0 + wv * 16;

  v16bf qf[4];
  {
    const bf16_t* qp = qbuf + (size_t)(s * SL + qrow0 + l16) * (NH * HD) + h * HD;
#pragma unroll
    for (int ch = 0; ch < 4; ++ch)
      qf[ch] = ld2x8(qp + 32 * ch + 8 * half, qp + 32 * ch + 16 + 8 * half);
  }

  v8f acc[8];
  for (int c = 0; c < 8; ++c)
    for (int r = 0; r < 8; ++r) acc[c][r] = 0.0f;
  float mrow[8], lrow[8];
  for (int r = 0; r < 8; ++r) { mrow[r] = -3.0e38f; lrow[r] = 0.0f; }

  const float scale = 0.08838834764831845f;  // 1/sqrt(128)
  const int nt = (q0 + 128) / 32;            // causal tile count for this slab

#define SLOT_BASE(T) (block_offsets[s * (SL / BS) + (((T) * 32) >> 6)] * BS + (((T) * 32) & (BS - 1)))

#if HAVE_TDM
  if (wv == 0) {  // prologue: tile 0 in flight
    int sb = SLOT_BASE(0);
    tdm_load_2d(kbuf + (size_t)sb * (NKV * HD) + g * HD,
                (unsigned)(uintptr_t)&Kt[0][0][0],
                128, 32, NKV * HD, NKV * HD, STOK, /*pad 64dw*/5, /*+4dw*/4);
    tdm_load_2d(vt + (size_t)g * HD * STOK + sb,
                (unsigned)(uintptr_t)&VtT[0][0][0],
                32, 128, STOK, STOK, NKV * HD, 3, 4);
  }
#endif
  int cur = 0;
  for (int t = 0; t < nt; ++t) {
    const int kt = t * 32;
#if HAVE_TDM
    if (wv == 0) {
      if (t + 1 < nt) {  // overlap: DMA tile t+1 while computing tile t
        int sb2 = SLOT_BASE(t + 1);
        tdm_load_2d(kbuf + (size_t)sb2 * (NKV * HD) + g * HD,
                    (unsigned)(uintptr_t)&Kt[1 - cur][0][0],
                    128, 32, NKV * HD, NKV * HD, STOK, 5, 4);
        tdm_load_2d(vt + (size_t)g * HD * STOK + sb2,
                    (unsigned)(uintptr_t)&VtT[1 - cur][0][0],
                    32, 128, STOK, STOK, NKV * HD, 3, 4);
        __builtin_amdgcn_s_wait_tensorcnt(2);
      } else {
        __builtin_amdgcn_s_wait_tensorcnt(0);
      }
    }
    __syncthreads();
#else
    {
      const int sb = SLOT_BASE(t);
      for (int i = tid; i < 512; i += 256) {
        int r = i >> 4, c8 = (i & 15) * 8;
        *(uint4*)&Kt[cur][r][c8] =
            *(const uint4*)(kbuf + (size_t)(sb + r) * (NKV * HD) + g * HD + c8);
      }
      int d = tid >> 1, ko = (tid & 1) * 16;
      *(uint4*)&VtT[cur][d][ko] =
          *(const uint4*)(vt + ((size_t)g * HD + d) * STOK + sb + ko);
    }
    __syncthreads();
#endif

    if (kt <= qrow0 + 15) {
      v8f d0, d1;
      for (int r = 0; r < 8; ++r) { d0[r] = 0.0f; d1[r] = 0.0f; }
#pragma unroll
      for (int ch = 0; ch < 4; ++ch) {
        const bf16_t* k0p = &Kt[cur][l16][32 * ch + 8 * half];
        const bf16_t* k1p = &Kt[cur][16 + l16][32 * ch + 8 * half];
        v16bf b0 = ld2x8(k0p, k0p + 16);
        v16bf b1 = ld2x8(k1p, k1p + 16);
        d0 = __builtin_amdgcn_wmma_f32_16x16x32_bf16(
            false, qf[ch], false, b0, (short)0, d0, false, false);
        d1 = __builtin_amdgcn_wmma_f32_16x16x32_bf16(
            false, qf[ch], false, b1, (short)0, d1, false, false);
      }

#pragma unroll
      for (int r = 0; r < 8; ++r) {
        int qi = qrow0 + r + 8 * half;
        float v0 = d0[r] * scale;
        float v1 = d1[r] * scale;
        if (kt + l16 > qi)      v0 = -1.0e30f;
        if (kt + 16 + l16 > qi) v1 = -1.0e30f;
        float mx = fmaxf(v0, v1);
#pragma unroll
        for (int o = 1; o < 16; o <<= 1) mx = fmaxf(mx, __shfl_xor(mx, o, 16));
        float mi    = fmaxf(mrow[r], mx);
        float alpha = __expf(mrow[r] - mi);
        float p0 = __expf(v0 - mi);
        float p1 = __expf(v1 - mi);
        float rs = p0 + p1;
#pragma unroll
        for (int o = 1; o < 16; o <<= 1) rs += __shfl_xor(rs, o, 16);
        lrow[r] = lrow[r] * alpha + rs;
        mrow[r] = mi;
#pragma unroll
        for (int c = 0; c < 8; ++c) acc[c][r] *= alpha;
        Pt[wv][r + 8 * half][l16]      = f2bf(p0);
        Pt[wv][r + 8 * half][16 + l16] = f2bf(p1);
      }

      const bf16_t* prow = &Pt[wv][l16][0];
      v16bf pa = ld2x8(prow + 8 * half, prow + 16 + 8 * half);
#pragma unroll
      for (int c = 0; c < 8; ++c) {
        const bf16_t* vrow = &VtT[cur][16 * c + l16][0];
        v16bf bv = ld2x8(vrow + 8 * half, vrow + 16 + 8 * half);
        acc[c] = __builtin_amdgcn_wmma_f32_16x16x32_bf16(
            false, pa, false, bv, (short)0, acc[c], false, false);
      }
    }
    __syncthreads();  // tile reads done before DMA re-targets this buffer
    cur ^= 1;
  }
#undef SLOT_BASE

#pragma unroll
  for (int c = 0; c < 8; ++c)
#pragma unroll
    for (int r = 0; r < 8; ++r) {
      int row = s * SL + qrow0 + r + 8 * half;
      ctxb[(size_t)row * (NH * HD) + h * HD + 16 * c + l16] = f2bf(acc[c][r] / lrow[r]);
    }
}

// ---------------------------------------------------------------------------
// Host launcher. Workspace (~219 MB):
//   r0: f32 QKV (75.5MB), reused as bf16 ctx after rope
//   r1: bf16 hidden [M][K]      r2: bf16 w_qkv^T [N][K]
//   r3: bf16 w_dense^T [N][K]   r4: bf16 Q (rope-applied)
//   r5: bf16 K cache [slot][g][d]  r6: bf16 V cache [g][d][slot]
// ---------------------------------------------------------------------------
extern "C" void kernel_launch(void* const* d_in, const int* in_sizes, int n_in,
                              void* d_out, int out_size, void* d_ws, size_t ws_size,
                              hipStream_t stream) {
  (void)in_sizes; (void)n_in; (void)out_size; (void)ws_size;
  const float* hidden  = (const float*)d_in[0];
  const float* rope    = (const float*)d_in[1];
  const float* w_qkv   = (const float*)d_in[2];
  const float* b_qkv   = (const float*)d_in[3];
  const float* w_dense = (const float*)d_in[4];
  const int*   blk_off = (const int*)d_in[7];
  float* out = (float*)d_out;

  char* ws = (char*)d_ws;
  size_t off = 0;
  float*  qkv  = (float*)(ws + off);  off += (size_t)STOK * QKV_N * 4;   // r0
  bf16_t* ctxb = (bf16_t*)qkv;                                           // alias r0
  bf16_t* hbf  = (bf16_t*)(ws + off); off += (size_t)STOK * HID * 2;     // r1
  bf16_t* wqT  = (bf16_t*)(ws + off); off += (size_t)QKV_N * HID * 2;    // r2
  bf16_t* wdT  = (bf16_t*)(ws + off); off += (size_t)HID * HID * 2;      // r3
  bf16_t* qbuf = (bf16_t*)(ws + off); off += (size_t)STOK * NH * HD * 2; // r4
  bf16_t* kbuf = (bf16_t*)(ws + off); off += (size_t)STOK * NKV * HD * 2;// r5
  bf16_t* vt   = (bf16_t*)(ws + off);                                    // r6

  // one-shot bf16 conversion / transposition (weights then live in L2)
  conv_bf16<<<(STOK * HID) / 1024, 256, 0, stream>>>(hidden, hbf);
  conv_bf16_T<<<dim3(QKV_N / 64, HID / 32), 256, 0, stream>>>(w_qkv, wqT, HID, QKV_N);
  conv_bf16_T<<<dim3(HID / 64, HID / 32), 256, 0, stream>>>(w_dense, wdT, HID, HID);

  // 1) fused QKV projection + bias
  gemm_bf16wmma<<<dim3(QKV_N / 128, STOK / 128), 256, 0, stream>>>(
      hbf, wqT, b_qkv, qkv, STOK, QKV_N, HID);
  // 2) RoPE + paged KV scatter
  rope_scatter<<<STOK, 256, 0, stream>>>(qkv, rope, blk_off, qbuf, kbuf, vt);
  // 3) causal GQA flash attention -> bf16 ctx (aliases r0)
  attention<<<NS * NH * (SL / 128), 256, 0, stream>>>(qbuf, kbuf, vt, blk_off, ctxb);
  // 4) dense output projection
  gemm_bf16wmma<<<dim3(HID / 128, STOK / 128), 256, 0, stream>>>(
      ctxb, wdT, nullptr, out, STOK, HID, NH * HD);
}
